// Interpolation_8924942041976
// MI455X (gfx1250) — matrix operand order repaired and checked by reading, not compiled, not run on hardware
//
#include <hip/hip_runtime.h>

typedef __attribute__((ext_vector_type(8)))  __bf16 v8bf;
typedef __attribute__((ext_vector_type(16))) __bf16 v16bf;
typedef __attribute__((ext_vector_type(8)))  float  v8f;

union Frag16 { v16bf v; v8bf h[2]; };

__device__ __forceinline__ unsigned short f2bf(float f) {
  union { float f; unsigned u; } v; v.f = f;
  unsigned u = v.u;
  u += 0x7FFFu + ((u >> 16) & 1u);   // round-to-nearest-even
  return (unsigned short)(u >> 16);
}
__device__ __forceinline__ float bf2f(unsigned short h) {
  union { unsigned u; float f; } v; v.u = ((unsigned)h) << 16;
  return v.f;
}

// Async global->LDS 128-bit copy (CDNA5 GLOBAL_LOAD_ASYNC_TO_LDS_B128,
// tracked by ASYNCcnt). ldsByteOff is relative to the workgroup LDS base
// (dynamic LDS starts at offset 0 here).
__device__ __forceinline__ void async_g2l_b128(unsigned ldsByteOff,
                                               const void* gsrc) {
#if defined(__gfx1250__)
  asm volatile("global_load_async_to_lds_b128 %0, %1, off"
               :: "v"(ldsByteOff), "v"(gsrc) : "memory");
#else
  (void)ldsByteOff; (void)gsrc;
#endif
}
__device__ __forceinline__ void wait_async_zero() {
#if defined(__gfx1250__)
  asm volatile("s_wait_asynccnt 0x0" ::: "memory");
#endif
}

// -------------------------------------------------------------------------
// Stage 0: 2x bilinear upsample (align_corners), NCHW fp32 -> NHWC bf16
// u layout: [4][256][256][64] bf16
// -------------------------------------------------------------------------
__global__ void upsample_kernel(const float* __restrict__ x,
                                unsigned short* __restrict__ u) {
  int n = blockIdx.x * 256 + threadIdx.x;   // 4*256*256*64 total
  int c = n & 63; int p = n >> 6;
  int xo = p & 255; p >>= 8;
  int yo = p & 255; int b = p >> 8;
  const float s = 127.0f / 255.0f;
  float py = yo * s, px = xo * s;
  int iy = (int)py, ix = (int)px;
  float fy = py - (float)iy, fx = px - (float)ix;
  int iy1 = min(iy + 1, 127), ix1 = min(ix + 1, 127);
  const float* xb = x + (size_t)(b * 64 + c) * 128 * 128;
  float a00 = xb[iy * 128 + ix],  a01 = xb[iy * 128 + ix1];
  float a10 = xb[iy1 * 128 + ix], a11 = xb[iy1 * 128 + ix1];
  float v = (1.f - fy) * ((1.f - fx) * a00 + fx * a01)
          +        fy  * ((1.f - fx) * a10 + fx * a11);
  u[n] = f2bf(v);
}

// -------------------------------------------------------------------------
// Stage 1: 3x3 conv + bias + ReLU as implicit GEMM on WMMA bf16.
// src/dst: [4][256][256][64] bf16 channel-last (channels >= Ci are zero).
// Wave w: ntile = w&3 (fixed co group), 8 output rows in 4 pairs.
// Input halo tile staged with async global->LDS b128 copies.
// -------------------------------------------------------------------------
__global__ void conv3x3_wmma(const unsigned short* __restrict__ src,
                             const float* __restrict__ W,
                             const float* __restrict__ Bias,
                             unsigned short* __restrict__ dst,
                             int Ci) {
  extern __shared__ __align__(32) unsigned short smem[];
  unsigned short* inT = smem;                // [18][18][64] bf16  (41.5 KB)
  unsigned short* wT  = smem + 18 * 18 * 64; // [64][576] bf16    (73.7 KB)
  int tid = threadIdx.x;
  int b = blockIdx.z;
  int y0 = blockIdx.y * 16, x0 = blockIdx.x * 16;

  // input halo tile: async copy in-image lanes, zero-fill halo lanes
  for (int idx = tid; idx < 18 * 18 * 8; idx += 256) {
    int p = idx >> 3, part = idx & 7;               // part = 8 channels
    int gy = y0 - 1 + p / 18, gx = x0 - 1 + p % 18;
    if (gy >= 0 && gy < 256 && gx >= 0 && gx < 256) {
      const unsigned short* gp =
          src + (((size_t)(b * 256 + gy) * 256 + gx) * 64 + part * 8);
      async_g2l_b128((unsigned)(p * 64 + part * 8) * 2u, gp);
    } else {
      *(uint4*)(inT + (size_t)p * 64 + part * 8) = make_uint4(0u, 0u, 0u, 0u);
    }
  }
  // weight matrix, K = tap*64 + ci, zero pad ci>=Ci and co>=51
  for (int idx = tid; idx < 64 * 576; idx += 256) {
    int co = idx / 576, kk = idx % 576;
    int tap = kk >> 6, ci = kk & 63;
    unsigned short v = 0;
    if (co < 51 && ci < Ci) v = f2bf(W[(size_t)(co * Ci + ci) * 9 + tap]);
    wT[idx] = v;
  }
  wait_async_zero();
  __syncthreads();

  int wave = tid >> 5, lane = tid & 31;
  int lhalf = lane >> 4;
  int l15 = lane & 15;
  int ntile = wave & 3;              // co group fixed per wave
  int mbase = (wave >> 2) * 8;       // 8 output rows per wave
  int co = ntile * 16 + l15;
  float bias = (co < 51) ? Bias[co] : 0.f;
  const unsigned short* wrow = wT + (size_t)co * 576 + lhalf * 16;

#pragma unroll 1
  for (int pr = 0; pr < 4; ++pr) {
    int m0 = mbase + pr * 2;         // pair of output rows m0, m0+1
    v8f d0 = {0.f, 0.f, 0.f, 0.f, 0.f, 0.f, 0.f, 0.f};
    v8f d1 = {0.f, 0.f, 0.f, 0.f, 0.f, 0.f, 0.f, 0.f};
#pragma unroll 1
    for (int tap = 0; tap < 9; ++tap) {
      int dy = tap / 3, dx = tap - dy * 3;
      Frag16 b0, b1, a0, a1, a2, a3;
      const unsigned short* wp = wrow + tap * 64;
      b0.v = *(const v16bf*)(wp);          // K chunk 0
      b1.v = *(const v16bf*)(wp + 32);     // K chunk 1
      const unsigned short* ar0 =
          inT + ((size_t)(m0 + dy) * 18 + (l15 + dx)) * 64 + lhalf * 16;
      const unsigned short* ar1 = ar0 + 18 * 64;   // next output row
      a0.h[0] = *(const v8bf*)(ar0);       a0.h[1] = *(const v8bf*)(ar0 + 8);
      a1.h[0] = *(const v8bf*)(ar0 + 32);  a1.h[1] = *(const v8bf*)(ar0 + 40);
      a2.h[0] = *(const v8bf*)(ar1);       a2.h[1] = *(const v8bf*)(ar1 + 8);
      a3.h[0] = *(const v8bf*)(ar1 + 32);  a3.h[1] = *(const v8bf*)(ar1 + 40);
      d0 = __builtin_amdgcn_wmma_f32_16x16x32_bf16(false, a0.v, false, b0.v,
                                                   (short)0, d0, false, false);
      d0 = __builtin_amdgcn_wmma_f32_16x16x32_bf16(false, a1.v, false, b1.v,
                                                   (short)0, d0, false, false);
      d1 = __builtin_amdgcn_wmma_f32_16x16x32_bf16(false, a2.v, false, b0.v,
                                                   (short)0, d1, false, false);
      d1 = __builtin_amdgcn_wmma_f32_16x16x32_bf16(false, a3.v, false, b1.v,
                                                   (short)0, d1, false, false);
    }
    // bias + relu + store both rows (C/D layout: M = r (+8), N = lane)
    size_t rowbase = ((size_t)(b * 256 + (y0 + m0)) * 256 + x0) * 64 + co;
#pragma unroll
    for (int r = 0; r < 8; ++r) {
      int px = r + lhalf * 8;
      float v0 = fmaxf(d0[r] + bias, 0.f);
      float v1 = fmaxf(d1[r] + bias, 0.f);
      if (co >= 51) { v0 = 0.f; v1 = 0.f; }   // keep pad channels zero
      dst[rowbase + (size_t)px * 64]            = f2bf(v0);
      dst[rowbase + (size_t)px * 64 + 256 * 64] = f2bf(v1);
    }
  }
}

// -------------------------------------------------------------------------
// Stage 2: adaptive separable conv via banded WMMA matmul.
//   A[(c,i), x'] = img[b,c,y+i,x0+x']      (153x96, bf16, zero padded)
//   B[x', x]     = kh[x'-x, y, x0+x]       (96x32 banded, bf16)
//   H = A*B  via 10 Mtiles x 2 Ntiles x 3 Kchunks of wmma_f32_16x16x32_bf16
//   out[c,x]     = sum_i kv[i,x] * H[(c,i),x]   (LDS ds_add_f32 reduction)
// -------------------------------------------------------------------------
__global__ void sepconv_wmma(const float* __restrict__ i1,
                             const float* __restrict__ i2,
                             const unsigned short* __restrict__ k1v,
                             const unsigned short* __restrict__ k1h,
                             const unsigned short* __restrict__ k2v,
                             const unsigned short* __restrict__ k2h,
                             float* __restrict__ out) {
  extern __shared__ __align__(32) unsigned short smem[];
  unsigned short* A   = smem;                    // [2][160][96] bf16
  unsigned short* Bc  = A + 2 * 160 * 96;        // [2][32][96]  bf16
  unsigned short* kvL = Bc + 2 * 32 * 96;        // [2][32][64]  bf16
  float* acc = (float*)(kvL + 2 * 32 * 64);      // [3][32] f32
  int tid = threadIdx.x;
  int b = blockIdx.z, y = blockIdx.y, x0 = blockIdx.x * 32;

  if (tid < 96) acc[tid] = 0.f;

  // kv: raw bf16 copy, async global->LDS (8-channel b128 chunks)
  const unsigned kvBase = (unsigned)(2 * 160 * 96 + 2 * 32 * 96) * 2u;
  for (int idx = tid; idx < 2 * 32 * 8; idx += 256) {
    int im = idx / (32 * 8), rem = idx % (32 * 8);
    int n = rem >> 3, part = rem & 7;
    const unsigned short* kv = im ? k2v : k1v;
    const unsigned short* gp =
        kv + ((size_t)(b * 256 + y) * 256 + (x0 + n)) * 64 + part * 8;
    async_g2l_b128(kvBase + (unsigned)((im * 32 + n) * 64 + part * 8) * 2u, gp);
  }
  // A: image rows (fp32 -> bf16), zero pad m>=153, k>=82
  for (int idx = tid; idx < 2 * 160 * 96; idx += 256) {
    int im = idx / (160 * 96), rem = idx % (160 * 96);
    int m = rem / 96, k = rem % 96;
    unsigned short v = 0;
    if (m < 153 && k < 82) {
      int c = m / 51, i = m - c * 51;
      const float* img = im ? i2 : i1;
      v = f2bf(img[((size_t)(b * 3 + c) * 306 + (y + i)) * 306 + (x0 + k)]);
    }
    A[idx] = v;
  }
  // B: banded horizontal-kernel matrix, column-major [px][k]
  for (int idx = tid; idx < 2 * 32 * 96; idx += 256) {
    int im = idx / (32 * 96), rem = idx % (32 * 96);
    int n = rem / 96, k = rem % 96;
    int j = k - n;
    unsigned short v = 0;
    if (j >= 0 && j < 51) {
      const unsigned short* kh = im ? k2h : k1h;
      v = kh[((size_t)(b * 256 + y) * 256 + (x0 + n)) * 64 + j];
    }
    Bc[idx] = v;
  }
  wait_async_zero();
  __syncthreads();

  int wave = tid >> 5, lane = tid & 31;
  int lhalf = lane >> 4;
  int l15 = lane & 15;

#pragma unroll 1
  for (int q = 0; q < 5; ++q) {
    int tile = wave + (q << 3);            // 40 tiles: 2 img x 10 M x 2 N
    int im = tile / 20, t2 = tile % 20;
    int mtile = t2 >> 1, ntile = t2 & 1;
    const unsigned short* Ai = A + (size_t)im * 160 * 96;
    const unsigned short* Bi = Bc + (size_t)im * 32 * 96;
    int px = ntile * 16 + l15;
    v8f d = {0.f, 0.f, 0.f, 0.f, 0.f, 0.f, 0.f, 0.f};
#pragma unroll
    for (int ch = 0; ch < 3; ++ch) {
      Frag16 a, bb;
      int row = mtile * 16 + l15;
      int kbA = ch * 32 + lhalf * 16;
      a.h[0] = *(const v8bf*)(Ai + (size_t)row * 96 + kbA);
      a.h[1] = *(const v8bf*)(Ai + (size_t)row * 96 + kbA + 8);
      int kbB = ch * 32 + lhalf * 16;
      bb.v = *(const v16bf*)(Bi + (size_t)px * 96 + kbB);
      d = __builtin_amdgcn_wmma_f32_16x16x32_bf16(
          false, a.v, false, bb.v, (short)0, d, false, false);
    }
    // vertical contraction: group per channel, then one LDS float atomic
    const unsigned short* kvi = kvL + ((size_t)im * 32 + px) * 64;
    int mbase = mtile * 16 + lhalf * 8;
    float partial = 0.f; int curc = -1;
#pragma unroll
    for (int r = 0; r < 8; ++r) {
      int m = mbase + r;
      if (m < 153) {
        int c = m / 51, i = m - c * 51;
        float t = d[r] * bf2f(kvi[i]);
        if (c != curc) {
          if (curc >= 0) atomicAdd(&acc[curc * 32 + px], partial);
          curc = c; partial = t;
        } else {
          partial += t;
        }
      }
    }
    if (curc >= 0) atomicAdd(&acc[curc * 32 + px], partial);
  }
  __syncthreads();
  if (tid < 96) {
    int c = tid / 32, n = tid & 31;
    out[((size_t)(b * 3 + c) * 256 + y) * 256 + (x0 + n)] = acc[c * 32 + n];
  }
}

// -------------------------------------------------------------------------
extern "C" void kernel_launch(void* const* d_in, const int* in_sizes, int n_in,
                              void* d_out, int out_size, void* d_ws, size_t ws_size,
                              hipStream_t stream) {
  const float* x  = (const float*)d_in[0];
  const float* i1 = (const float*)d_in[1];
  const float* i2 = (const float*)d_in[2];
  const float* W1 = (const float*)d_in[3];
  const float* B1 = (const float*)d_in[4];
  const float* W2 = (const float*)d_in[5];
  const float* B2 = (const float*)d_in[6];
  const float* W3 = (const float*)d_in[7];
  const float* B3 = (const float*)d_in[8];
  float* out = (float*)d_out;

  const size_t plane = (size_t)4 * 256 * 256 * 64;   // bf16 elements
  unsigned short* u  = (unsigned short*)d_ws;        // upsampled features
  unsigned short* hA = u + plane;                    // conv scratch
  unsigned short* hB = hA + plane;
  unsigned short* kb = hB + plane;                   // 4 predicted kernel maps

  upsample_kernel<<<dim3((unsigned)(plane / 256)), 256, 0, stream>>>(x, u);

  dim3 cgrid(16, 16, 4);
  size_t convLds = (size_t)(18 * 18 * 64 + 64 * 576) * sizeof(unsigned short);
  for (int k = 0; k < 4; ++k) {
    conv3x3_wmma<<<cgrid, 256, convLds, stream>>>(
        u, W1 + (size_t)k * 51 * 64 * 9, B1 + k * 51, hA, 64);
    conv3x3_wmma<<<cgrid, 256, convLds, stream>>>(
        hA, W2 + (size_t)k * 51 * 51 * 9, B2 + k * 51, hB, 51);
    conv3x3_wmma<<<cgrid, 256, convLds, stream>>>(
        hB, W3 + (size_t)k * 51 * 51 * 9, B3 + k * 51, kb + (size_t)k * plane, 51);
  }

  size_t sepLds = (size_t)(2 * 160 * 96 + 2 * 32 * 96 + 2 * 32 * 64)
                      * sizeof(unsigned short) + 96 * sizeof(float);
  sepconv_wmma<<<dim3(8, 256, 4), 256, sepLds, stream>>>(
      i1, i2, kb + 0 * plane, kb + 1 * plane, kb + 2 * plane, kb + 3 * plane, out);
}
